// CalculateAttention_36215164240130
// MI455X (gfx1250) — matrix operand order repaired and checked by reading, not compiled, not run on hardware
//
#include <hip/hip_runtime.h>
#include <hip/hip_bf16.h>

typedef __attribute__((ext_vector_type(16))) __bf16 v16bf;
typedef __attribute__((ext_vector_type(8)))  __bf16 v8bf;
typedef __attribute__((ext_vector_type(8)))  float  v8f;

#define BATCH 4
#define HEADS 16
#define SEQ   2048
#define DH    64
#define NW    8            // waves per workgroup
#define QB    (NW * 16)    // 128 query rows per workgroup
#define KC    64           // key-chunk per iteration
#define KCP   72           // padded row stride (halfs) for transposed V tile

// exp2-domain scale: softmax(s/8) => p = 2^((s - m) * 0.125 * log2(e))
#define EXP2_SCALE 0.180336880f

__global__ __launch_bounds__(256)
void fa_fwd_kernel(const float* __restrict__ Qg, const float* __restrict__ Kg,
                   const float* __restrict__ Vg, const unsigned char* __restrict__ Mg,
                   float* __restrict__ Og)
{
    __shared__ __align__(32) __bf16 sK[KC * DH];         //  8 KB, [key][d]
    __shared__ __align__(32) __bf16 sV[DH * KCP];        //  9 KB, [d][key] transposed, padded
    __shared__ __align__(32) __bf16 sP[NW * 16 * KC];    // 16 KB, per-wave P scratch

    const int tid  = threadIdx.x;
    const int wave = tid >> 5;
    const int lane = tid & 31;
    const int half = lane >> 4;    // 0: lanes 0-15, 1: lanes 16-31
    const int l16  = lane & 15;

    const int nQB   = SEQ / QB;                // 16
    const int bh    = blockIdx.x / nQB;        // b*H + h
    const int qblk  = blockIdx.x % nQB;
    const int b     = bh / HEADS;
    const int qbase = qblk * QB + wave * 16;

    const float* Qh = Qg + (size_t)bh * SEQ * DH;
    const float* Kh = Kg + (size_t)bh * SEQ * DH;
    const float* Vh = Vg + (size_t)bh * SEQ * DH;
    const unsigned char* Mb = Mg + (size_t)b * SEQ * SEQ;
    float* Oh = Og + (size_t)bh * SEQ * DH;

    // ---- Q rows as two 16x32 bf16 A-fragments (K = 0..31, 32..63)
    // A as loaded here, contiguous k order (16x32): lane%16 = row M; elem i -> k = (lane/16)*16 + i
    v16bf qa[2];
    #pragma unroll
    for (int f = 0; f < 2; ++f) {
        #pragma unroll
        for (int i = 0; i < 16; ++i) {
            int k = f * 32 + half * 16 + i;
            qa[f][i] = (__bf16)Qh[(size_t)(qbase + l16) * DH + k];
        }
    }

    // ---- Online-softmax state. C/D layout: row = r + 8*half, col = lane%16.
    v8f acc[4];
    #pragma unroll
    for (int n = 0; n < 4; ++n) acc[n] = (v8f){};
    float mrow[8], lrow[8];
    #pragma unroll
    for (int r = 0; r < 8; ++r) { mrow[r] = -3.0e38f; lrow[r] = 0.0f; }

    __bf16* pScratch = &sP[wave * 16 * KC];

    for (int kb = 0; kb < SEQ; kb += KC) {
        __syncthreads();
        // ---- Stage K (row-major bf16) and V (transposed bf16) chunks into LDS.
        // float4 global loads; K stored as packed bf16x4 (b64), V strided b16.
        #pragma unroll
        for (int i = 0; i < 4; ++i) {
            int q4 = i * 256 + tid;            // float4 index, 0..1023
            int kr = q4 >> 4;                  // key row (16 float4 per row)
            int d0 = (q4 & 15) * 4;
            float4 kv = *(const float4*)&Kh[(size_t)(kb + kr) * DH + d0];
            float4 vv = *(const float4*)&Vh[(size_t)(kb + kr) * DH + d0];
            union { __bf16 h[4]; uint2 u; } pk;
            pk.h[0] = (__bf16)kv.x; pk.h[1] = (__bf16)kv.y;
            pk.h[2] = (__bf16)kv.z; pk.h[3] = (__bf16)kv.w;
            *(uint2*)&sK[kr * DH + d0] = pk.u;
            sV[(d0 + 0) * KCP + kr] = (__bf16)vv.x;
            sV[(d0 + 1) * KCP + kr] = (__bf16)vv.y;
            sV[(d0 + 2) * KCP + kr] = (__bf16)vv.z;
            sV[(d0 + 3) * KCP + kr] = (__bf16)vv.w;
        }
        __syncthreads();

        // ---- Prefetch next chunk (global_prefetch_b8), 64B per thread covers 16KB
        if (kb + KC < SEQ) {
            __builtin_prefetch(&Kh[(size_t)(kb + KC) * DH + tid * 16], 0, 3);
            __builtin_prefetch(&Vh[(size_t)(kb + KC) * DH + tid * 16], 0, 3);
        }

        // ---- Scores: four 16x16 tiles, each = two K=32 WMMAs
        // B as loaded here, contiguous k order (32x16): lanes 0-15 hold K 0-15, lanes 16-31 hold K 16-31
        v8f s[4];
        #pragma unroll
        for (int t = 0; t < 4; ++t) {
            v8f c = (v8f){};
            #pragma unroll
            for (int f = 0; f < 2; ++f) {
                const v16bf* bp =
                    (const v16bf*)&sK[(t * 16 + l16) * DH + f * 32 + half * 16];
                c = __builtin_amdgcn_wmma_f32_16x16x32_bf16(
                        false, qa[f], false, *bp, (short)0, c, false, false);
            }
            s[t] = c;
        }

        // ---- Masked-fill (before scaling, like the reference), in raw-score domain
        float zm[4][8];
        #pragma unroll
        for (int t = 0; t < 4; ++t) {
            #pragma unroll
            for (int r = 0; r < 8; ++r) {
                int qrow = qbase + r + half * 8;
                unsigned char mk = Mb[(size_t)qrow * SEQ + kb + t * 16 + l16];
                zm[t][r] = mk ? -1.0e9f : s[t][r];
            }
        }

        // ---- Row max across 16 lanes of each half-group (raw domain; scale>0 is monotonic)
        float mnew[8];
        #pragma unroll
        for (int r = 0; r < 8; ++r) {
            float mx = fmaxf(fmaxf(zm[0][r], zm[1][r]), fmaxf(zm[2][r], zm[3][r]));
            #pragma unroll
            for (int off = 1; off < 16; off <<= 1)
                mx = fmaxf(mx, __shfl_xor(mx, off, 32));
            mnew[r] = fmaxf(mrow[r], mx);
        }

        float cf[8];
        #pragma unroll
        for (int r = 0; r < 8; ++r)
            cf[r] = exp2f((mrow[r] - mnew[r]) * EXP2_SCALE);

        // ---- P = 2^((s-m)*C): bf16 P to per-wave LDS scratch; row sums; update stats
        #pragma unroll
        for (int r = 0; r < 8; ++r) {
            float mC  = mnew[r] * EXP2_SCALE;
            int   row = r + half * 8;
            float ps  = 0.0f;
            #pragma unroll
            for (int t = 0; t < 4; ++t) {
                float p = exp2f(zm[t][r] * EXP2_SCALE - mC);
                pScratch[row * KC + t * 16 + l16] = (__bf16)p;
                ps += p;
            }
            #pragma unroll
            for (int off = 1; off < 16; off <<= 1)
                ps += __shfl_xor(ps, off, 32);
            lrow[r] = lrow[r] * cf[r] + ps;
            mrow[r] = mnew[r];
        }

        // ---- Rescale output accumulators
        #pragma unroll
        for (int n = 0; n < 4; ++n)
            #pragma unroll
            for (int r = 0; r < 8; ++r)
                acc[n][r] *= cf[r];

        // ---- P back as two A-fragments (K = 0..31, 32..63), contiguous 16B LDS reads
        v16bf pa[2];
        #pragma unroll
        for (int f = 0; f < 2; ++f) {
            const v8bf* lo = (const v8bf*)&pScratch[l16 * KC + f * 32 + half * 16];
            const v8bf* hi = (const v8bf*)&pScratch[l16 * KC + f * 32 + half * 16 + 8];
            v8bf a = *lo, bvec = *hi;
            #pragma unroll
            for (int i = 0; i < 8; ++i) { pa[f][i] = a[i]; pa[f][i + 8] = bvec[i]; }
        }

        // ---- O += P(16x64) @ V(64x64): 4 d-tiles x 2 K-halves = 8 WMMAs
        #pragma unroll
        for (int n = 0; n < 4; ++n) {
            #pragma unroll
            for (int f = 0; f < 2; ++f) {
                const v16bf* vb =
                    (const v16bf*)&sV[(n * 16 + l16) * KCP + f * 32 + half * 16];
                acc[n] = __builtin_amdgcn_wmma_f32_16x16x32_bf16(
                             false, pa[f], false, *vb, (short)0, acc[n], false, false);
            }
        }
    }

    // ---- Normalize and write out (fp32)
    #pragma unroll
    for (int r = 0; r < 8; ++r) {
        float inv = 1.0f / lrow[r];
        int qrow = qbase + r + half * 8;
        #pragma unroll
        for (int n = 0; n < 4; ++n)
            Oh[(size_t)qrow * DH + n * 16 + l16] = acc[n][r] * inv;
    }
}

extern "C" void kernel_launch(void* const* d_in, const int* in_sizes, int n_in,
                              void* d_out, int out_size, void* d_ws, size_t ws_size,
                              hipStream_t stream) {
    const float* Q = (const float*)d_in[0];
    const float* K = (const float*)d_in[1];
    const float* V = (const float*)d_in[2];
    const unsigned char* M = (const unsigned char*)d_in[3];  // bool mask, 1 byte/elt
    float* O = (float*)d_out;

    dim3 grid(BATCH * HEADS * (SEQ / QB));  // 64 * 16 = 1024 workgroups
    dim3 block(256);                        // 8 wave32 waves
    fa_fwd_kernel<<<grid, block, 0, stream>>>(Q, K, V, M, O);
}
